// LlamaSwiftKVAttention_65627100283444
// MI455X (gfx1250) — matrix operand order repaired and checked by reading, not compiled, not run on hardware
//
#include <hip/hip_runtime.h>
#include <hip/hip_bf16.h>
#include <stdint.h>

// ---------------------------------------------------------------------------
// CDNA5 (gfx1250) fused Llama attention block.
// bf16 WMMA everywhere (fp32 accumulate); all LDS staging is done with
// global_load_async_to_lds_b128 + ASYNCcnt double-buffering, straight-line
// staging with pointer increments (no per-slab address multiplies).
// ---------------------------------------------------------------------------

typedef __attribute__((ext_vector_type(16))) __bf16 v16bf;
typedef __attribute__((ext_vector_type(8)))  float  v8f;

union Frag { v16bf v; uint32_t u[8]; };

static __device__ __forceinline__ uint16_t f32_to_bf16(float f) {
    uint32_t x = __float_as_uint(f);
    uint32_t r = x + 0x7FFFu + ((x >> 16) & 1u);   // round-to-nearest-even
    return (uint16_t)(r >> 16);
}
static __device__ __forceinline__ uint32_t pack2(uint16_t lo, uint16_t hi) {
    return (uint32_t)lo | ((uint32_t)hi << 16);
}
static __device__ __forceinline__ void async_b128(uint32_t lds_dst, uint64_t gsrc) {
    asm volatile("global_load_async_to_lds_b128 %0, %1, off"
                 :: "v"(lds_dst), "v"(gsrc) : "memory");
}
#define ASYNC_WAIT_4() asm volatile("s_wait_asynccnt 4" ::: "memory")
#define ASYNC_WAIT_0() asm volatile("s_wait_asynccnt 0" ::: "memory")

// ---------------------------------------------------------------------------
// GEMM: C[M,N] fp32 = A[M,K](bf16, row-major) * BT[N,K](bf16, row-major = B^T)
// Block: 256 thr = 8 waves. Block tile 128x128, wave tile 32x64, K-slab 32.
// Pure async staging (4 b128/wave/slab), double-buffered LDS.
// ---------------------------------------------------------------------------
__global__ __launch_bounds__(256) void gemm_bf16_wmma(
    const uint16_t* __restrict__ A, const uint16_t* __restrict__ BT,
    float* __restrict__ C, int M, int N, int K)
{
    __shared__ uint32_t As[2][128][16];   // [buf][row][kpair]  8KB/buf
    __shared__ uint32_t Bs[2][128][16];   // [buf][col n][kpair]

    const int tid  = threadIdx.x;
    const int lane = tid & 31;
    const int wave = tid >> 5;
    const int hl   = lane & 15;
    const int hi   = (lane < 16) ? 0 : 1;

    const int row0 = blockIdx.y * 128;
    const int col0 = blockIdx.x * 128;
    const int rblk = (wave >> 1) * 32;    // 0,32,64,96
    const int cblk = (wave & 1) * 64;     // 0,64

    const uint32_t asb = (uint32_t)(uintptr_t)&As[0][0][0];
    const uint32_t bsb = (uint32_t)(uintptr_t)&Bs[0][0][0];

    // Per-thread async transfer descriptors: rows rA and rA+64, one 16B segment.
    const int rA  = tid >> 2;            // 0..63
    const int sgA = tid & 3;             // 16B segment within 64B row
    uint64_t srcA0 = (uint64_t)(uintptr_t)(A  + (size_t)(row0 + rA) * K + sgA * 8);
    uint64_t srcA1 = srcA0 + (uint64_t)64 * K * 2;     // +64 rows
    uint64_t srcB0 = (uint64_t)(uintptr_t)(BT + (size_t)(col0 + rA) * K + sgA * 8);
    uint64_t srcB1 = srcB0 + (uint64_t)64 * K * 2;
    const uint32_t dst0 = (uint32_t)(rA * 64 + sgA * 16);
    const uint32_t dst1 = dst0 + 64 * 64;

    v8f z = {};
    v8f acc[2][4];
#pragma unroll
    for (int ti = 0; ti < 2; ++ti)
#pragma unroll
        for (int tn = 0; tn < 4; ++tn) acc[ti][tn] = z;

    auto stage = [&](int b) {              // stages the *next* 32-k slab
        const uint32_t ab = asb + (uint32_t)b * 8192u;
        const uint32_t bb = bsb + (uint32_t)b * 8192u;
        async_b128(ab + dst0, srcA0);
        async_b128(ab + dst1, srcA1);
        async_b128(bb + dst0, srcB0);
        async_b128(bb + dst1, srcB1);
        srcA0 += 64; srcA1 += 64; srcB0 += 64; srcB1 += 64;   // 32 bf16 = 64 B
    };

    const int nslab = K >> 5;
    stage(0);
    for (int s = 0; s < nslab; ++s) {
        const int b = s & 1;
        if (s + 1 < nslab) { stage((s + 1) & 1); ASYNC_WAIT_4(); }
        else               { ASYNC_WAIT_0(); }
        __syncthreads();

        Frag a[2], bf[4];
#pragma unroll
        for (int ti = 0; ti < 2; ++ti)
#pragma unroll
            for (int i = 0; i < 4; ++i) {
                // A fragment: lane<16 kpairs {0..3,8..11}; lane>=16 {4..7,12..15}
                a[ti].u[i]     = As[b][rblk + ti * 16 + hl][i + hi * 4];
                a[ti].u[i + 4] = As[b][rblk + ti * 16 + hl][8 + i + hi * 4];
            }
#pragma unroll
        for (int tn = 0; tn < 4; ++tn)
#pragma unroll
            for (int i = 0; i < 8; ++i)   // B fragment: kpairs {0..3,8..11} / {4..7,12..15}
                bf[tn].u[i] = Bs[b][cblk + tn * 16 + hl][(i & 3) + (i >> 2) * 8 + hi * 4];

#pragma unroll
        for (int ti = 0; ti < 2; ++ti)
#pragma unroll
            for (int tn = 0; tn < 4; ++tn)
                acc[ti][tn] = __builtin_amdgcn_wmma_f32_16x16x32_bf16(
                    false, a[ti].v, false, bf[tn].v, (short)0, acc[ti][tn], false, false);
        __syncthreads();
    }

    // C/D layout: col = lane&15, row = j + (lane>=16 ? 8 : 0)
#pragma unroll
    for (int ti = 0; ti < 2; ++ti)
#pragma unroll
        for (int tn = 0; tn < 4; ++tn)
#pragma unroll
            for (int j = 0; j < 8; ++j) {
                int m = j + hi * 8;
                C[(size_t)(row0 + rblk + ti * 16 + m) * N + col0 + cblk + tn * 16 + hl]
                    = acc[ti][tn][j];
            }
}

// ---------------------------------------------------------------------------
// fp32 -> bf16 flat convert (hidden states).
// ---------------------------------------------------------------------------
__global__ void h_cvt_kernel(const float* __restrict__ x, uint16_t* __restrict__ y, int n)
{
    int i = blockIdx.x * blockDim.x + threadIdx.x;
    if (i < n) y[i] = f32_to_bf16(x[i]);
}

// ---------------------------------------------------------------------------
// Weight fp32[K,N] -> bf16 transposed WT[N,K], LDS-tiled (coalesced both ways).
// Grid (N/32, K/32), block 256 = (32,8) logical.
// ---------------------------------------------------------------------------
__global__ __launch_bounds__(256) void wcvt_transpose_kernel(
    const float* __restrict__ W, uint16_t* __restrict__ WT, int K, int N)
{
    __shared__ uint16_t tile[32][33];
    const int tx = threadIdx.x & 31, ty = threadIdx.x >> 5;
    const int n0 = blockIdx.x * 32, k0 = blockIdx.y * 32;
#pragma unroll
    for (int r = ty; r < 32; r += 8)
        tile[r][tx] = f32_to_bf16(W[(size_t)(k0 + r) * N + n0 + tx]);
    __syncthreads();
#pragma unroll
    for (int r = ty; r < 32; r += 8)
        WT[(size_t)(n0 + r) * K + k0 + tx] = tile[tx][r];
}

// ---------------------------------------------------------------------------
// RoPE on projected Q (fp32), emit bf16 Q in [S][NH*D] layout.
// ---------------------------------------------------------------------------
__global__ void rope_cvt_kernel(const float* __restrict__ qtmp,
                                const int* __restrict__ pos_ids,
                                uint16_t* __restrict__ qb)
{
    const int s = blockIdx.x, h = blockIdx.y, i = threadIdx.x;   // i in [0,64)
    const float pos = (float)pos_ids[s];
    const float freq = pos * __expf(-(float)i * (9.210340371976184f / 64.0f));
    const float c  = __cosf(freq);
    const float sn = __sinf(freq);
    const float* qr = qtmp + (size_t)s * 4096 + h * 128;
    const float q1 = qr[i], q2 = qr[i + 64];
    uint16_t* out = qb + (size_t)s * 4096 + h * 128;
    out[i]      = f32_to_bf16(q1 * c - q2 * sn);
    out[i + 64] = f32_to_bf16(q2 * c + q1 * sn);
}

// ---------------------------------------------------------------------------
// K -> bf16 row-major; V -> bf16 pair-transposed vbT[kvh][d][tpair] (u32).
// ---------------------------------------------------------------------------
__global__ void kv_cvt_kernel(const float* __restrict__ k, const float* __restrict__ v,
                              uint16_t* __restrict__ kb, uint32_t* __restrict__ vbT)
{
    int i = blockIdx.x * blockDim.x + threadIdx.x;          // < 1048576
    kb[2 * i]     = f32_to_bf16(k[2 * i]);
    kb[2 * i + 1] = f32_to_bf16(k[2 * i + 1]);
    int kvh = i >> 17, rem = i & 131071, d = rem >> 10, tp = rem & 1023;
    const float* vp = v + ((size_t)kvh * 2048 + tp * 2) * 128 + d;
    vbT[i] = pack2(f32_to_bf16(vp[0]), f32_to_bf16(vp[128]));
}

// ---------------------------------------------------------------------------
// Flash attention, causal, GQA (G=4). Grid (NH=32, S/128=16), 256 threads.
// Wave owns a 16-query tile. 32-key chunks, double-buffered async K/V in LDS.
// Scores computed transposed (S^T = K * Q^T) so probs -> P A-fragment is
// lane-local; softmax reductions need one shfl_xor(16). Output stored bf16.
// Fully-masked chunks skip compute per wave (contribution underflows to 0).
// ---------------------------------------------------------------------------
__global__ __launch_bounds__(256) void attn_kernel(
    const uint16_t* __restrict__ qb, const uint16_t* __restrict__ kb,
    const uint32_t* __restrict__ vbT, uint16_t* __restrict__ obb)
{
    __shared__ uint32_t Ks[2][32][64];    // [buf][key t][d pair]   8KB/buf
    __shared__ uint32_t Vs[2][128][16];   // [buf][d][t pair]       8KB/buf

    const int tid  = threadIdx.x;
    const int lane = tid & 31;
    const int wave = tid >> 5;
    const int hl   = lane & 15;
    const int hi   = (lane < 16) ? 0 : 1;

    const int h     = blockIdx.x;
    const int qblk  = blockIdx.y;
    const int kvh   = h >> 2;                 // G = NH/NKV = 4
    const int qbase = qblk * 128 + wave * 16;
    const int s_idx = qbase + hl;

    const uint32_t ksb = (uint32_t)(uintptr_t)&Ks[0][0][0];
    const uint32_t vsb = (uint32_t)(uintptr_t)&Vs[0][0][0];

    // Per-thread async transfer descriptors (2 K + 2 V per 32-key chunk).
    const int tK  = tid >> 4;            // 0..15 (key rows tK, tK+16)
    const int sgK = tid & 15;            // 16B segment within 256B row
    uint64_t srcK0 = (uint64_t)(uintptr_t)(kb + (((size_t)kvh * 2048 + tK) << 7) + sgK * 8);
    uint64_t srcK1 = srcK0 + 16 * 256;                 // +16 key rows
    const uint32_t dK0 = (uint32_t)(tK * 256 + sgK * 16);
    const uint32_t dK1 = dK0 + 16 * 256;
    const int dV  = tid >> 2;            // 0..63 (d rows dV, dV+64)
    const int sgV = tid & 3;             // 16B segment within 64B chunk-slice
    uint64_t srcV0 = (uint64_t)(uintptr_t)(vbT + (((size_t)kvh * 128 + dV) << 10) + sgV * 4);
    uint64_t srcV1 = srcV0 + (uint64_t)64 * 1024 * 4;  // +64 d rows
    const uint32_t dV0 = (uint32_t)(dV * 64 + sgV * 16);
    const uint32_t dV1 = dV0 + 64 * 64;

    // Q^T B-fragments (4 x 32-d slabs), loaded once; dpairs {0..3,8..11} / {4..7,12..15} per lane.
    Frag qf[4];
    {
        const uint32_t* qrow = (const uint32_t*)(qb + (size_t)s_idx * 4096 + h * 128);
#pragma unroll
        for (int c = 0; c < 4; ++c)
#pragma unroll
            for (int i = 0; i < 8; ++i)
                qf[c].u[i] = qrow[c * 16 + (i & 3) + (i >> 2) * 8 + hi * 4];
    }

    v8f z = {};
    v8f o[8];
#pragma unroll
    for (int n = 0; n < 8; ++n) o[n] = z;
    float m_i = -1e30f, l_i = 0.0f;
    const float scale = 0.08838834764831845f;   // 1/sqrt(128)

    auto stageKV = [&](int b) {            // stages the *next* 32-key chunk
        const uint32_t kbase = ksb + (uint32_t)b * 8192u;
        const uint32_t vbase = vsb + (uint32_t)b * 8192u;
        async_b128(kbase + dK0, srcK0);
        async_b128(kbase + dK1, srcK1);
        async_b128(vbase + dV0, srcV0);
        async_b128(vbase + dV1, srcV1);
        srcK0 += 32 * 256; srcK1 += 32 * 256;   // +32 keys x 256B
        srcV0 += 64;       srcV1 += 64;         // +16 tpairs x 4B
    };

    const int nchunk = (qblk + 1) * 4;          // causal pruning (block level)
    stageKV(0);
    for (int ch = 0; ch < nchunk; ++ch) {
        const int b  = ch & 1;
        const int t0 = ch * 32;
        if (ch + 1 < nchunk) { stageKV((ch + 1) & 1); ASYNC_WAIT_4(); }
        else                 { ASYNC_WAIT_0(); }
        __syncthreads();

        // Wave-level causal skip: chunk fully masked for this wave's queries.
        if (t0 <= qbase + 15) {
            // --- S^T tiles: rows = keys, cols = queries ---
            v8f st[2];
#pragma unroll
            for (int kt = 0; kt < 2; ++kt) {
                v8f acc = z;
                const int trow = kt * 16 + hl;
#pragma unroll
                for (int c = 0; c < 4; ++c) {
                    Frag a;
#pragma unroll
                    for (int i = 0; i < 4; ++i) {
                        a.u[i]     = Ks[b][trow][c * 16 + i + hi * 4];
                        a.u[i + 4] = Ks[b][trow][c * 16 + 8 + i + hi * 4];
                    }
                    acc = __builtin_amdgcn_wmma_f32_16x16x32_bf16(
                            false, a.v, false, qf[c].v, (short)0, acc, false, false);
                }
                st[kt] = acc;
            }

            // --- mask + online softmax (query = column = hl) ---
            float sc[16];
            float mc = -1e30f;
#pragma unroll
            for (int kt = 0; kt < 2; ++kt)
#pragma unroll
                for (int j = 0; j < 8; ++j) {
                    int t_idx = t0 + kt * 16 + j + hi * 8;
                    float v = st[kt][j] * scale;
                    v = (t_idx > s_idx) ? -10000.0f : v;
                    sc[kt * 8 + j] = v;
                    mc = fmaxf(mc, v);
                }
            mc = fmaxf(mc, __shfl_xor(mc, 16, 32));
            const float m_new = fmaxf(m_i, mc);
            const float alpha = __expf(m_i - m_new);

            Frag pf;
            float lsum = 0.0f;
#pragma unroll
            for (int i = 0; i < 16; i += 2) {
                float p0 = __expf(sc[i]     - m_new);
                float p1 = __expf(sc[i + 1] - m_new);
                lsum += p0 + p1;
                pf.u[i >> 1] = pack2(f32_to_bf16(p0), f32_to_bf16(p1));
            }
            lsum += __shfl_xor(lsum, 16, 32);
            l_i = l_i * alpha + lsum;
            m_i = m_new;

            // rescale running O (row m = j + hi*8; alpha for query m in lane m)
#pragma unroll
            for (int j = 0; j < 8; ++j) {
                float aj = __shfl(alpha, j + hi * 8, 32);
#pragma unroll
                for (int n = 0; n < 8; ++n) o[n][j] *= aj;
            }

            // --- P * V : A = P (lane-local from st tiles), B = V from LDS ---
#pragma unroll
            for (int n = 0; n < 8; ++n) {
                Frag bv;
                const int drow = n * 16 + hl;
#pragma unroll
                for (int i = 0; i < 8; ++i) bv.u[i] = Vs[b][drow][(i & 3) + (i >> 2) * 8 + hi * 4];
                o[n] = __builtin_amdgcn_wmma_f32_16x16x32_bf16(
                         false, pf.v, false, bv.v, (short)0, o[n], false, false);
            }
        }
        __syncthreads();
    }

    // --- normalize, convert to bf16, store: obb[s][h*128 + d] ---
    const float rinv = 1.0f / l_i;
#pragma unroll
    for (int j = 0; j < 8; ++j) {
        float rj = __shfl(rinv, j + hi * 8, 32);
        int srow = qbase + j + hi * 8;
        uint16_t* orow = obb + (size_t)srow * 4096 + h * 128;
#pragma unroll
        for (int n = 0; n < 8; ++n) orow[n * 16 + hl] = f32_to_bf16(o[n][j] * rj);
    }
}

// ---------------------------------------------------------------------------
// Host-side orchestration (graph-capture safe: only kernel launches on stream)
// ---------------------------------------------------------------------------
extern "C" void kernel_launch(void* const* d_in, const int* in_sizes, int n_in,
                              void* d_out, int out_size, void* d_ws, size_t ws_size,
                              hipStream_t stream)
{
    (void)in_sizes; (void)n_in; (void)out_size; (void)ws_size;

    const float* hidden = (const float*)d_in[0];   // [1,2048,4096]
    const float* kc     = (const float*)d_in[1];   // [1,8,2048,128]
    const float* vc     = (const float*)d_in[2];   // [1,8,2048,128]
    const float* Wq     = (const float*)d_in[3];   // [4096,4096]
    const float* Wo     = (const float*)d_in[4];   // [4096,4096]
    const int*   pos    = (const int*)d_in[5];     // [1,2048]
    // d_in[6] attention_mask: causal, implemented analytically in attn_kernel.
    float* out = (float*)d_out;                    // [1,2048,4096] fp32

    // Workspace (88 MB), regions reused under stream ordering:
    //  R1 16MB: hb (bf16 hidden)  -> reused as qb (bf16 rotated Q)
    //  R2 32MB: WqT (bf16)        -> reused as WoT
    //  R3 32MB: qtmp (fp32 Q)     -> reused as obb (bf16 attention out)
    //  R4  4MB: kb  (bf16 K, row-major)
    //  R5  4MB: vbT (bf16 V, pair-transposed u32)
    char* w = (char*)d_ws;
    uint16_t* hb   = (uint16_t*)w;                                   // R1
    uint16_t* qbuf = hb;                                             // R1 alias
    uint16_t* wT   = (uint16_t*)(w + (size_t)16 * 1024 * 1024);      // R2
    float*    qtmp = (float*)   (w + (size_t)48 * 1024 * 1024);      // R3
    uint16_t* obb  = (uint16_t*)qtmp;                                // R3 alias
    uint16_t* kb   = (uint16_t*)(w + (size_t)80 * 1024 * 1024);      // R4
    uint32_t* vbT  = (uint32_t*)(w + (size_t)84 * 1024 * 1024);      // R5

    // 1) hidden -> bf16
    h_cvt_kernel<<<32768, 256, 0, stream>>>(hidden, hb, 2048 * 4096);
    // 2) WqT = transpose(Wq) in bf16
    wcvt_transpose_kernel<<<dim3(128, 128), 256, 0, stream>>>(Wq, wT, 4096, 4096);
    // 3) Q projection: qtmp = hidden @ Wq   (WMMA, async double-buffered)
    gemm_bf16_wmma<<<dim3(32, 16), 256, 0, stream>>>(hb, wT, qtmp, 2048, 4096, 4096);
    // 4) RoPE + bf16 convert (R1 reuse: hb is dead)
    rope_cvt_kernel<<<dim3(2048, 32), 64, 0, stream>>>(qtmp, pos, qbuf);
    // 5) K/V -> bf16 (K row-major, V pair-transposed)
    kv_cvt_kernel<<<4096, 256, 0, stream>>>(kc, vc, kb, vbT);
    // 6) causal GQA flash attention -> obb (R3 reuse: qtmp is dead)
    attn_kernel<<<dim3(32, 16), 256, 0, stream>>>(qbuf, kb, vbT, obb);
    // 7) WoT = transpose(Wo) in bf16 (R2 reuse: WqT is dead)
    wcvt_transpose_kernel<<<dim3(128, 128), 256, 0, stream>>>(Wo, wT, 4096, 4096);
    // 8) Output projection: out = obb @ Wo
    gemm_bf16_wmma<<<dim3(32, 16), 256, 0, stream>>>(obb, wT, out, 2048, 4096, 4096);
}
